// MusicAttrSingleGMVAE_25632364822946
// MI455X (gfx1250) — compile-verified, matrix operand order checked by reading, not run on hardware
//
#include <hip/hip_runtime.h>
#include <hip/hip_bf16.h>

#define B_   256
#define S_   256
#define R_   342
#define RP_  352
#define H_   1024
#define Z_   256
#define NK_  4
#define H3_  (3 * H_)
#define LOG2PI_ 1.8378770664093453f

typedef __attribute__((ext_vector_type(16))) __bf16 v16bf;
typedef __attribute__((ext_vector_type(8)))  __bf16 v8bf;
typedef __attribute__((ext_vector_type(8)))  float  v8f;

__device__ __forceinline__ __bf16 f2bf(float f) {
  union { float f; unsigned u; } v; v.f = f;
  unsigned r = (v.u + 0x7FFFu + ((v.u >> 16) & 1u)) >> 16;
  union { unsigned short s; __bf16 b; } o; o.s = (unsigned short)r;
  return o.b;
}

__device__ __forceinline__ float sigm(float x) { return 1.0f / (1.0f + __expf(-x)); }

// ---------------------------------------------------------------------------
// Generic bf16 WMMA GEMM:  C[M,N] = A[M,K] @ W[N,K]^T + bias[N]
// A, W row-major bf16 (contiguous along K).  One wave32 per 16x16 C tile.
// Fragment layouts follow the CDNA5 ISA 16-bit A/B VGPR layout exactly.
// ---------------------------------------------------------------------------
__global__ __launch_bounds__(128)
void k_gemm(const __bf16* __restrict__ A, int lda,
            const __bf16* __restrict__ W, int ldw,
            const float* __restrict__ bias,
            float* __restrict__ C, int ldc,
            __bf16* __restrict__ Cbf, int ldcbf,
            int M, int N, int Kd)
{
  const int lane = threadIdx.x & 31;
  const int wave = threadIdx.x >> 5;
  const int tn = blockIdx.x * 4 + wave;   // N tile index
  const int tm = blockIdx.y;              // M tile index
  if (tn * 16 >= N) return;               // uniform per wave

  const int mrow = tm * 16 + (lane & 15);
  const int nrow = tn * 16 + (lane & 15);
  const int koff = (lane >> 4) * 8;       // lane-half K offset per ISA layout

  const __bf16* pa = A + (size_t)mrow * lda + koff;
  const __bf16* pw = W + (size_t)nrow * ldw + koff;

  v8f acc = {};
  for (int k = 0; k < Kd; k += 32) {
    __builtin_prefetch((const void*)(pw + 256), 0, 1);   // global_prefetch_b8 on weight stream
    v8bf a0 = *(const v8bf*)(pa);
    v8bf a1 = *(const v8bf*)(pa + 16);
    v8bf b0 = *(const v8bf*)(pw);
    v8bf b1 = *(const v8bf*)(pw + 16);
    v16bf av = __builtin_shufflevector(a0, a1, 0,1,2,3,4,5,6,7,8,9,10,11,12,13,14,15);
    v16bf bv = __builtin_shufflevector(b0, b1, 0,1,2,3,4,5,6,7,8,9,10,11,12,13,14,15);
    acc = __builtin_amdgcn_wmma_f32_16x16x32_bf16(
        /*neg_a=*/false, av, /*neg_b=*/false, bv,
        /*c_mod=*/(short)0, acc, /*reuse_a=*/false, /*reuse_b=*/false);
    pa += 32; pw += 32;
  }

  // Epilogue per documented 32-bit C/D layout: col = lane&15, row = v + (lane>>4)*8
  const int ccol = tn * 16 + (lane & 15);
  const int crow0 = tm * 16 + (lane >> 4) * 8;
  const float bb = bias ? bias[ccol] : 0.0f;
#pragma unroll
  for (int v = 0; v < 8; ++v) {
    float val = acc[v] + bb;
    C[(size_t)(crow0 + v) * ldc + ccol] = val;
    if (Cbf) Cbf[(size_t)(crow0 + v) * ldcbf + ccol] = f2bf(val);
  }
}

// ---------------------------------------------------------------------------
// Weight prep: fp32 -> bf16 with optional column-window extraction + zero pad
// dst[r*dcols+c] = (r<srows && c<scols) ? src[r*sld + scol0 + c] : 0
// ---------------------------------------------------------------------------
__global__ void k_prep_w(const float* __restrict__ src, int sld, int scol0,
                         int srows, int scols,
                         __bf16* __restrict__ dst, int drows, int dcols)
{
  int i = blockIdx.x * blockDim.x + threadIdx.x;
  if (i >= drows * dcols) return;
  int r = i / dcols, c = i % dcols;
  float v = (r < srows && c < scols) ? src[(size_t)r * sld + scol0 + c] : 0.0f;
  dst[i] = f2bf(v);
}

__global__ void k_pad_bias(const float* __restrict__ src, float* __restrict__ dst,
                           int n, int np)
{
  int i = blockIdx.x * blockDim.x + threadIdx.x;
  if (i >= np) return;
  dst[i] = (i < n) ? src[i] : 0.0f;
}

// x [B,S,R] fp32 -> xbf [S,B,RP] bf16 (K padded with zeros)
__global__ void k_cast_x(const float* __restrict__ x, __bf16* __restrict__ xbf)
{
  int i = blockIdx.x * blockDim.x + threadIdx.x;
  if (i >= S_ * B_ * RP_) return;
  int c = i % RP_; int sb = i / RP_;
  int b = sb % B_; int s = sb / B_;
  float v = (c < R_) ? x[((size_t)b * S_ + s) * R_ + c] : 0.0f;
  xbf[i] = f2bf(v);
}

__global__ void k_init_idx(int* idx) {
  int i = blockIdx.x * blockDim.x + threadIdx.x;
  if (i < B_) idx[i] = R_ - 1;   // out0 one-hot at last position
}

// Standard GRU cell update (gi, gh kept separate for the n-gate).
__global__ void k_gru_update(const float* __restrict__ gi, const float* __restrict__ gh,
                             const float* __restrict__ hprev,
                             float* __restrict__ h, __bf16* __restrict__ hbf)
{
  int i = blockIdx.x * blockDim.x + threadIdx.x;
  if (i >= B_ * H_) return;
  int b = i / H_, j = i % H_;
  const size_t o = (size_t)b * H3_;
  float r  = sigm(gi[o + j] + gh[o + j]);
  float zg = sigm(gi[o + H_ + j] + gh[o + H_ + j]);
  float n  = tanhf(gi[o + 2 * H_ + j] + r * gh[o + 2 * H_ + j]);
  float hp = hprev[i];
  float hn = (1.0f - zg) * n + zg * hp;
  h[i] = hn;
  hbf[i] = f2bf(hn);
}

// Decoder cell A update: gi = z_gi (precomputed) + one-hot column gather of cg_w_ih.
__global__ void k_upd_dec_a(const float* __restrict__ zgi, const float* __restrict__ gh,
                            const float* __restrict__ wih /* [3H, Z+R] fp32 */,
                            const int* __restrict__ idx,
                            float* __restrict__ h, __bf16* __restrict__ hbf)
{
  int i = blockIdx.x * blockDim.x + threadIdx.x;
  if (i >= B_ * H_) return;
  int b = i / H_, j = i % H_;
  const int ld = Z_ + R_;
  const int col = idx[b];
  const size_t o = (size_t)b * H3_;
  float gir = zgi[o + j]            + wih[(size_t)j * ld + col];
  float giz = zgi[o + H_ + j]       + wih[(size_t)(H_ + j) * ld + col];
  float gin = zgi[o + 2 * H_ + j]   + wih[(size_t)(2 * H_ + j) * ld + col];
  float r  = sigm(gir + gh[o + j]);
  float zg = sigm(giz + gh[o + H_ + j]);
  float n  = tanhf(gin + r * gh[o + 2 * H_ + j]);
  float hp = h[i];
  float hn = (1.0f - zg) * n + zg * hp;
  h[i] = hn;
  hbf[i] = f2bf(hn);
}

// enc_bf[b, 0:H]=h_fwd[b]; enc_bf[b, H:2H]=h_bwd[b]
__global__ void k_concat_enc(const float* __restrict__ h, __bf16* __restrict__ enc)
{
  int i = blockIdx.x * blockDim.x + threadIdx.x;
  if (i >= B_ * 2 * H_) return;
  int b = i / (2 * H_), j = i % (2 * H_);
  float v = (j < H_) ? h[(size_t)b * H_ + j]
                     : h[(size_t)(B_ + b) * H_ + (j - H_)];
  enc[i] = f2bf(v);
}

// std = exp(varlin); z = mu + std*eps; mirror z to ws fp32 + bf16
__global__ void k_std_z(const float* __restrict__ varlin, const float* __restrict__ mu,
                        const float* __restrict__ eps,
                        float* __restrict__ std_out, float* __restrict__ z_out,
                        float* __restrict__ z_f, __bf16* __restrict__ z_bf)
{
  int i = blockIdx.x * blockDim.x + threadIdx.x;
  if (i >= B_ * Z_) return;
  float sd = __expf(varlin[i]);
  float zz = mu[i] + sd * eps[i];
  std_out[i] = sd; z_out[i] = zz; z_f[i] = zz; z_bf[i] = f2bf(zz);
}

// GMM posterior: llh over K=4 components, softmax, argmax
__global__ __launch_bounds__(256)
void k_gmm(const float* __restrict__ z, const float* __restrict__ mu_lk,
           const float* __restrict__ lv_lk,
           float* __restrict__ logLogit, float* __restrict__ qy, float* __restrict__ y)
{
  int b = blockIdx.x * blockDim.x + threadIdx.x;
  if (b >= B_) return;
  float ll[NK_];
  for (int k = 0; k < NK_; ++k) {
    float s = 0.0f;
    for (int zz = 0; zz < Z_; ++zz) {
      float d  = z[(size_t)b * Z_ + zz] - mu_lk[(size_t)k * Z_ + zz];
      float lv = lv_lk[(size_t)k * Z_ + zz];
      s += d * d * __expf(-lv) + lv + LOG2PI_;
    }
    ll[k] = -0.5f * s + __logf(1.0f / NK_);
  }
  float m = ll[0]; int mi = 0;
  for (int k = 1; k < NK_; ++k) if (ll[k] > m) { m = ll[k]; mi = k; }
  float e[NK_], se = 0.0f;
  for (int k = 0; k < NK_; ++k) { e[k] = __expf(ll[k] - m); se += e[k]; }
  for (int k = 0; k < NK_; ++k) {
    logLogit[b * NK_ + k] = ll[k];
    qy[b * NK_ + k] = e[k] / se;
  }
  y[b] = (float)mi;
}

// log_softmax over 342 logits + argmax feedback index. One wave32 per row.
__global__ __launch_bounds__(256)
void k_logsoftmax(const float* __restrict__ lg, float* __restrict__ out, int s,
                  int* __restrict__ idxbuf)
{
  int wave = threadIdx.x >> 5;
  int lane = threadIdx.x & 31;
  int b = blockIdx.x * 8 + wave;
  if (b >= B_) return;
  const float* row = lg + (size_t)b * RP_;
  float mx = -3.0e38f; int mi = 0;
  for (int r = lane; r < R_; r += 32) {
    float v = row[r];
    if (v > mx) { mx = v; mi = r; }
  }
  for (int off = 16; off; off >>= 1) {
    float om = __shfl_xor(mx, off, 32);
    int   oi = __shfl_xor(mi, off, 32);
    if (om > mx || (om == mx && oi < mi)) { mx = om; mi = oi; }
  }
  float se = 0.0f;
  for (int r = lane; r < R_; r += 32) se += __expf(row[r] - mx);
  for (int off = 16; off; off >>= 1) se += __shfl_xor(se, off, 32);
  float lse = __logf(se) + mx;
  float* orow = out + (size_t)b * S_ * R_ + (size_t)s * R_;
  for (int r = lane; r < R_; r += 32) orow[r] = row[r] - lse;
  if (lane == 0) idxbuf[b] = mi;
}

// ---------------------------------------------------------------------------
extern "C" void kernel_launch(void* const* d_in, const int* in_sizes, int n_in,
                              void* d_out, int out_size, void* d_ws, size_t ws_size,
                              hipStream_t stream)
{
  (void)in_sizes; (void)n_in; (void)out_size; (void)ws_size;
  const float* x        = (const float*)d_in[0];
  const float* eps      = (const float*)d_in[1];
  const float* w_ih_f   = (const float*)d_in[2];
  const float* w_hh_f   = (const float*)d_in[3];
  const float* b_ih_f   = (const float*)d_in[4];
  const float* b_hh_f   = (const float*)d_in[5];
  const float* w_ih_b   = (const float*)d_in[6];
  const float* w_hh_b   = (const float*)d_in[7];
  const float* b_ih_b   = (const float*)d_in[8];
  const float* b_hh_b   = (const float*)d_in[9];
  const float* mu_w     = (const float*)d_in[10];
  const float* mu_b     = (const float*)d_in[11];
  const float* var_w    = (const float*)d_in[12];
  const float* var_b    = (const float*)d_in[13];
  const float* mu_lk    = (const float*)d_in[14];
  const float* lv_lk    = (const float*)d_in[15];
  const float* lig_w    = (const float*)d_in[16];
  const float* lig_b    = (const float*)d_in[17];
  const float* cg_w_ih  = (const float*)d_in[18];
  const float* cg_w_hh  = (const float*)d_in[19];
  const float* cg_b_ih  = (const float*)d_in[20];
  const float* cg_b_hh  = (const float*)d_in[21];
  const float* cg2_w_ih = (const float*)d_in[22];
  const float* cg2_w_hh = (const float*)d_in[23];
  const float* cg2_b_ih = (const float*)d_in[24];
  const float* cg2_b_hh = (const float*)d_in[25];
  const float* lo_w     = (const float*)d_in[26];
  const float* lo_b     = (const float*)d_in[27];

  float* out        = (float*)d_out;
  float* out_output = out;
  float* out_mu     = out + (size_t)B_ * S_ * R_;
  float* out_std    = out_mu  + (size_t)B_ * Z_;
  float* out_z      = out_std + (size_t)B_ * Z_;
  float* out_ll     = out_z   + (size_t)B_ * Z_;
  float* out_qy     = out_ll  + (size_t)B_ * NK_;
  float* out_y      = out_qy  + (size_t)B_ * NK_;

  // ---- workspace bump allocator ----
  char* wsp = (char*)d_ws;
  size_t off = 0;
  auto alloc = [&](size_t bytes) -> void* {
    void* p = wsp + off;
    off = (off + bytes + 255) & ~(size_t)255;
    return p;
  };
  __bf16* xbf       = (__bf16*)alloc((size_t)S_ * B_ * RP_ * 2);
  __bf16* wihf_bf   = (__bf16*)alloc((size_t)H3_ * RP_ * 2);
  __bf16* wihb_bf   = (__bf16*)alloc((size_t)H3_ * RP_ * 2);
  __bf16* whhf_bf   = (__bf16*)alloc((size_t)H3_ * H_ * 2);
  __bf16* whhb_bf   = (__bf16*)alloc((size_t)H3_ * H_ * 2);
  __bf16* muw_bf    = (__bf16*)alloc((size_t)Z_ * 2 * H_ * 2);
  __bf16* varw_bf   = (__bf16*)alloc((size_t)Z_ * 2 * H_ * 2);
  __bf16* ligw_bf   = (__bf16*)alloc((size_t)H_ * Z_ * 2);
  __bf16* wz_bf     = (__bf16*)alloc((size_t)H3_ * Z_ * 2);       // cg_w_ih z-columns
  __bf16* cgwhh_bf  = (__bf16*)alloc((size_t)H3_ * H_ * 2);
  __bf16* cg2wih_bf = (__bf16*)alloc((size_t)H3_ * H_ * 2);
  __bf16* cg2whh_bf = (__bf16*)alloc((size_t)H3_ * H_ * 2);
  __bf16* lowp_bf   = (__bf16*)alloc((size_t)RP_ * H_ * 2);
  float*  lobp_f    = (float*)alloc((size_t)RP_ * 4);
  float*  h_enc_f   = (float*)alloc((size_t)2 * B_ * H_ * 4);     // [dir, B, H]
  __bf16* h_enc_bf  = (__bf16*)alloc((size_t)2 * B_ * H_ * 2);
  float*  g_ih      = (float*)alloc((size_t)2 * B_ * H3_ * 4);
  float*  g_hh      = (float*)alloc((size_t)2 * B_ * H3_ * 4);
  __bf16* enc_bf    = (__bf16*)alloc((size_t)B_ * 2 * H_ * 2);
  float*  varlin    = (float*)alloc((size_t)B_ * Z_ * 4);
  float*  z_f       = (float*)alloc((size_t)B_ * Z_ * 4);
  __bf16* z_bf      = (__bf16*)alloc((size_t)B_ * Z_ * 2);
  float*  z_gi      = (float*)alloc((size_t)B_ * H3_ * 4);
  float*  h_a_f     = (float*)alloc((size_t)B_ * H_ * 4);
  __bf16* h_a_bf    = (__bf16*)alloc((size_t)B_ * H_ * 2);
  float*  h_b2_f    = (float*)alloc((size_t)B_ * H_ * 4);
  __bf16* h_b2_bf   = (__bf16*)alloc((size_t)B_ * H_ * 2);
  float*  g_ah      = (float*)alloc((size_t)B_ * H3_ * 4);
  float*  g2i       = (float*)alloc((size_t)B_ * H3_ * 4);
  float*  g2h       = (float*)alloc((size_t)B_ * H3_ * 4);
  float*  logits    = (float*)alloc((size_t)B_ * RP_ * 4);
  int*    idx       = (int*)alloc((size_t)B_ * 4);

  auto prep = [&](const float* src, int sld, int scol0, int srows, int scols,
                  __bf16* dst, int drows, int dcols) {
    int n = drows * dcols;
    k_prep_w<<<(n + 255) / 256, 256, 0, stream>>>(src, sld, scol0, srows, scols,
                                                  dst, drows, dcols);
  };
  auto gemm = [&](const __bf16* A, int lda, const __bf16* W, int ldw,
                  const float* bias, float* C, int ldc,
                  __bf16* Cbf, int ldcbf, int M, int N, int Kd) {
    dim3 g((N + 63) / 64, M / 16);
    k_gemm<<<g, 128, 0, stream>>>(A, lda, W, ldw, bias, C, ldc, Cbf, ldcbf, M, N, Kd);
  };

  // ---- weight / input prep (all on stream) ----
  prep(w_ih_f,  R_,       0,  H3_, R_,      wihf_bf,   H3_, RP_);
  prep(w_ih_b,  R_,       0,  H3_, R_,      wihb_bf,   H3_, RP_);
  prep(w_hh_f,  H_,       0,  H3_, H_,      whhf_bf,   H3_, H_);
  prep(w_hh_b,  H_,       0,  H3_, H_,      whhb_bf,   H3_, H_);
  prep(mu_w,    2 * H_,   0,  Z_,  2 * H_,  muw_bf,    Z_,  2 * H_);
  prep(var_w,   2 * H_,   0,  Z_,  2 * H_,  varw_bf,   Z_,  2 * H_);
  prep(lig_w,   Z_,       0,  H_,  Z_,      ligw_bf,   H_,  Z_);
  prep(cg_w_ih, Z_ + R_,  R_, H3_, Z_,      wz_bf,     H3_, Z_);   // z-part only
  prep(cg_w_hh, H_,       0,  H3_, H_,      cgwhh_bf,  H3_, H_);
  prep(cg2_w_ih,H_,       0,  H3_, H_,      cg2wih_bf, H3_, H_);
  prep(cg2_w_hh,H_,       0,  H3_, H_,      cg2whh_bf, H3_, H_);
  prep(lo_w,    H_,       0,  R_,  H_,      lowp_bf,   RP_, H_);   // pad N rows
  k_pad_bias<<<2, 256, 0, stream>>>(lo_b, lobp_f, R_, RP_);
  {
    int n = S_ * B_ * RP_;
    k_cast_x<<<(n + 255) / 256, 256, 0, stream>>>(x, xbf);
  }
  hipMemsetAsync(h_enc_f,  0, (size_t)2 * B_ * H_ * 4, stream);
  hipMemsetAsync(h_enc_bf, 0, (size_t)2 * B_ * H_ * 2, stream);
  k_init_idx<<<1, 256, 0, stream>>>(idx);

  const int upd_grid = (B_ * H_ + 255) / 256;

  // ---- encoder: 256 sequential steps, fwd+bwd directions in flight together ----
  for (int t = 0; t < S_; ++t) {
    const __bf16* xf = xbf + (size_t)t * B_ * RP_;
    const __bf16* xb = xbf + (size_t)(S_ - 1 - t) * B_ * RP_;
    gemm(xf, RP_, wihf_bf, RP_, b_ih_f, g_ih,                    H3_, nullptr, 0, B_, H3_, RP_);
    gemm(xb, RP_, wihb_bf, RP_, b_ih_b, g_ih + (size_t)B_ * H3_, H3_, nullptr, 0, B_, H3_, RP_);
    gemm(h_enc_bf,                    H_, whhf_bf, H_, b_hh_f, g_hh,                    H3_, nullptr, 0, B_, H3_, H_);
    gemm(h_enc_bf + (size_t)B_ * H_,  H_, whhb_bf, H_, b_hh_b, g_hh + (size_t)B_ * H3_, H3_, nullptr, 0, B_, H3_, H_);
    k_gru_update<<<upd_grid, 256, 0, stream>>>(g_ih, g_hh, h_enc_f, h_enc_f, h_enc_bf);
    k_gru_update<<<upd_grid, 256, 0, stream>>>(g_ih + (size_t)B_ * H3_, g_hh + (size_t)B_ * H3_,
                                               h_enc_f + (size_t)B_ * H_,
                                               h_enc_f + (size_t)B_ * H_,
                                               h_enc_bf + (size_t)B_ * H_);
  }

  // ---- latent heads ----
  {
    int n = B_ * 2 * H_;
    k_concat_enc<<<(n + 255) / 256, 256, 0, stream>>>(h_enc_f, enc_bf);
  }
  gemm(enc_bf, 2 * H_, muw_bf,  2 * H_, mu_b,  out_mu, Z_, nullptr, 0, B_, Z_, 2 * H_);
  gemm(enc_bf, 2 * H_, varw_bf, 2 * H_, var_b, varlin, Z_, nullptr, 0, B_, Z_, 2 * H_);
  {
    int n = B_ * Z_;
    k_std_z<<<(n + 255) / 256, 256, 0, stream>>>(varlin, out_mu, eps, out_std, out_z, z_f, z_bf);
  }
  k_gmm<<<1, 256, 0, stream>>>(z_f, mu_lk, lv_lk, out_ll, out_qy, out_y);

  // ---- decoder init: z-dependent gate input computed ONCE; hg0 = z@lig_w.T + lig_b ----
  gemm(z_bf, Z_, wz_bf,   Z_, cg_b_ih, z_gi,  H3_, nullptr, 0,  B_, H3_, Z_);
  gemm(z_bf, Z_, ligw_bf, Z_, lig_b,   h_a_f, H_,  h_a_bf,  H_, B_, H_,  Z_);

  // ---- decoder: 256 sequential steps; one-hot feedback folded into a gather ----
  for (int t = 0; t < S_; ++t) {
    // cell A: gates_h = h_a @ cg_w_hh.T + cg_b_hh
    gemm(h_a_bf, H_, cgwhh_bf, H_, cg_b_hh, g_ah, H3_, nullptr, 0, B_, H3_, H_);
    k_upd_dec_a<<<upd_grid, 256, 0, stream>>>(z_gi, g_ah, cg_w_ih, idx, h_a_f, h_a_bf);

    // cell B: gi2 = h_a @ cg2_w_ih.T ; gh2 = hprev @ cg2_w_hh.T (hprev = h_a at t==0)
    gemm(h_a_bf, H_, cg2wih_bf, H_, cg2_b_ih, g2i, H3_, nullptr, 0, B_, H3_, H_);
    const __bf16* hprev_bf = (t == 0) ? h_a_bf : h_b2_bf;
    const float*  hprev_f  = (t == 0) ? h_a_f  : h_b2_f;
    gemm(hprev_bf, H_, cg2whh_bf, H_, cg2_b_hh, g2h, H3_, nullptr, 0, B_, H3_, H_);
    k_gru_update<<<upd_grid, 256, 0, stream>>>(g2i, g2h, hprev_f, h_b2_f, h_b2_bf);

    // output projection + log_softmax + argmax feedback
    gemm(h_b2_bf, H_, lowp_bf, H_, lobp_f, logits, RP_, nullptr, 0, B_, RP_, H_);
    k_logsoftmax<<<B_ / 8, 256, 0, stream>>>(logits, out_output, t, idx);
  }
}
